// ClassifyLayer_21706764714528
// MI455X (gfx1250) — matrix operand order repaired and checked by reading, not compiled, not run on hardware
//
#include <hip/hip_runtime.h>
#include <hip/hip_bf16.h>

// ---------------------------------------------------------------------------
// Problem constants (fixed by setup_inputs: N=8192, D=512, len_prd=4096, head=1)
// ---------------------------------------------------------------------------
#define N_TOTAL 8192
#define DIM     512
#define LEN     4096          // len_prd  -> rows of output
#define NCOLS   4096          // N_TOTAL - len_prd -> cols of output
#define NEGV    (-99.0f)
#define EPSV    (1e-9f)

typedef __attribute__((ext_vector_type(16))) _Float16 v16h;
typedef __attribute__((ext_vector_type(8)))  _Float16 v8h;
typedef __attribute__((ext_vector_type(8)))  float    v8f;
typedef __attribute__((ext_vector_type(4)))  float    v4f;

// ---------------------------------------------------------------------------
// Pass 1: row-normalize embeddings (f32), emit f16 row-major into workspace.
// One wave32 per row; each lane owns 16 contiguous elements (512/32).
// ---------------------------------------------------------------------------
__global__ __launch_bounds__(256) void normalize_to_f16(
    const float* __restrict__ emb, _Float16* __restrict__ ne) {
  const int row  = blockIdx.x * 8 + (threadIdx.x >> 5);
  const int lane = threadIdx.x & 31;
  const float* p = emb + (size_t)row * DIM + lane * 16;

  v4f x0 = *(const v4f*)(p + 0);
  v4f x1 = *(const v4f*)(p + 4);
  v4f x2 = *(const v4f*)(p + 8);
  v4f x3 = *(const v4f*)(p + 12);

  float s = 0.0f;
#pragma unroll
  for (int i = 0; i < 4; ++i) s += x0[i] * x0[i];
#pragma unroll
  for (int i = 0; i < 4; ++i) s += x1[i] * x1[i];
#pragma unroll
  for (int i = 0; i < 4; ++i) s += x2[i] * x2[i];
#pragma unroll
  for (int i = 0; i < 4; ++i) s += x3[i] * x3[i];

  // wave32 butterfly reduction
#pragma unroll
  for (int off = 16; off >= 1; off >>= 1) s += __shfl_xor(s, off, 32);

  float nrm = sqrtf(s);
  nrm = (nrm <= EPSV) ? EPSV : nrm;
  const float inv = 1.0f / nrm;

  v8h lo, hi;
#pragma unroll
  for (int i = 0; i < 4; ++i) { lo[i]     = (_Float16)(x0[i] * inv);
                                lo[i + 4] = (_Float16)(x1[i] * inv);
                                hi[i]     = (_Float16)(x2[i] * inv);
                                hi[i + 4] = (_Float16)(x3[i] * inv); }

  _Float16* q = ne + (size_t)row * DIM + lane * 16;
  *(v8h*)(q)     = lo;   // global_store_b128
  *(v8h*)(q + 8) = hi;   // global_store_b128
}

// ---------------------------------------------------------------------------
// Pass 2: NT GEMM via v_wmma_f32_16x16x32_f16, 2x2 register-blocked per wave
// (32x32 wave tile, 4 accumulators) with fused epilogue:
//   sim = ne[:4096] @ ne[4096:]^T ; masked = sim - 99*(1-W);
//   out1 = clamp(masked,0,1) ; ext[..col] = 1/(1-masked)
// Block = 256 threads = 8 waves laid out 4(M) x 2(N) -> 128x64 block tile.
// K loop: 512/32 = 16 steps, 4 WMMA per step, 2 b128 loads per WMMA.
// ---------------------------------------------------------------------------
__global__ __launch_bounds__(256) void gemm_wmma_epilogue(
    const _Float16* __restrict__ ne, const int* __restrict__ Wmat,
    float* __restrict__ out1, float* __restrict__ ext) {
  const int wv     = threadIdx.x >> 5;
  const int lane   = threadIdx.x & 31;
  const int hi     = lane >> 4;        // 0: lanes 0-15, 1: lanes 16-31
  const int lr     = lane & 15;

  const int waveM  = wv >> 1;          // 0..3
  const int waveN  = wv & 1;           // 0..1
  const int tileRow = blockIdx.y * 128 + waveM * 32;  // in [0, 4096)
  const int tileCol = blockIdx.x * 64  + waveN * 32;  // in [0, 4096)

  // A fragment sources: rows (tileRow + lr) and (tileRow + 16 + lr) of ne[:4096]
  const _Float16* arow0 = ne + (size_t)(tileRow + lr) * DIM;
  const _Float16* arow1 = arow0 + (size_t)16 * DIM;
  // B fragment sources: rows (4096 + tileCol [+16] + lr) of ne (columns of B)
  const _Float16* brow0 = ne + (size_t)(LEN + tileCol + lr) * DIM;
  const _Float16* brow1 = brow0 + (size_t)16 * DIM;

  // Prefetch this lane's W-mask rows while the WMMAs run (global_prefetch_b8);
  // one touch per output row: the fetched line covers the 32-column span.
#pragma unroll
  for (int tm = 0; tm < 2; ++tm) {
#pragma unroll
    for (int r = 0; r < 8; ++r) {
      const int gr = tileRow + tm * 16 + r + 8 * hi;
      __builtin_prefetch(Wmat + (size_t)gr * N_TOTAL + LEN + tileCol + lr, 0, 0);
    }
  }

  v8f c00 = {}, c01 = {}, c10 = {}, c11 = {};
#pragma unroll 2
  for (int kb = 0; kb < DIM; kb += 32) {
    // A 16x32 f16 fragments: lane half selects K chunk parity (ISA A layout)
    v8h a0lo = *(const v8h*)(arow0 + kb + 8 * hi);
    v8h a0hi = *(const v8h*)(arow0 + kb + 16 + 8 * hi);
    v8h a1lo = *(const v8h*)(arow1 + kb + 8 * hi);
    v8h a1hi = *(const v8h*)(arow1 + kb + 16 + 8 * hi);
    v16h a0 = __builtin_shufflevector(a0lo, a0hi, 0, 1, 2, 3, 4, 5, 6, 7,
                                      8, 9, 10, 11, 12, 13, 14, 15);
    v16h a1 = __builtin_shufflevector(a1lo, a1hi, 0, 1, 2, 3, 4, 5, 6, 7,
                                      8, 9, 10, 11, 12, 13, 14, 15);
    // B 32x16 f16 fragments: same K chunk layout as A (8*hi and 16+8*hi)
    v8h b0lo = *(const v8h*)(brow0 + kb + 8 * hi);
    v8h b0hi = *(const v8h*)(brow0 + kb + 16 + 8 * hi);
    v8h b1lo = *(const v8h*)(brow1 + kb + 8 * hi);
    v8h b1hi = *(const v8h*)(brow1 + kb + 16 + 8 * hi);
    v16h b0 = __builtin_shufflevector(b0lo, b0hi, 0, 1, 2, 3, 4, 5, 6, 7,
                                      8, 9, 10, 11, 12, 13, 14, 15);
    v16h b1 = __builtin_shufflevector(b1lo, b1hi, 0, 1, 2, 3, 4, 5, 6, 7,
                                      8, 9, 10, 11, 12, 13, 14, 15);

    // 8 args: (neg_a, A, neg_b, B, c_mod, C, reuse_a, reuse_b)
    c00 = __builtin_amdgcn_wmma_f32_16x16x32_f16(false, a0, false, b0, (short)0, c00, false, false);
    c01 = __builtin_amdgcn_wmma_f32_16x16x32_f16(false, a0, false, b1, (short)0, c01, false, false);
    c10 = __builtin_amdgcn_wmma_f32_16x16x32_f16(false, a1, false, b0, (short)0, c10, false, false);
    c11 = __builtin_amdgcn_wmma_f32_16x16x32_f16(false, a1, false, b1, (short)0, c11, false, false);
  }

  // Fused epilogue; C layout: VGPR r -> row (r + 8*hi), col lr
  const v8f acc[2][2] = {{c00, c01}, {c10, c11}};
#pragma unroll
  for (int tm = 0; tm < 2; ++tm) {
#pragma unroll
    for (int tn = 0; tn < 2; ++tn) {
      const v8f c = acc[tm][tn];
#pragma unroll
      for (int r = 0; r < 8; ++r) {
        const int gr = tileRow + tm * 16 + r + 8 * hi;
        const int gc = tileCol + tn * 16 + lr;
        const float w = (float)Wmat[(size_t)gr * N_TOTAL + LEN + gc];
        const float m = c[r] + NEGV * (1.0f - w);
        out1[(size_t)gr * NCOLS + gc] = fminf(fmaxf(m, 0.0f), 1.0f);
        ext[(size_t)gr * (NCOLS + 1) + gc] = 1.0f / (1.0f - m);
      }
    }
  }
}

// ---------------------------------------------------------------------------
// Pass 3: per-row sum-normalization of `half` (stored in ext) + supple flag.
// supple = 1 iff all out1[row,:] == 0  (out1 > 0  <=>  masked > 0).
// ---------------------------------------------------------------------------
__global__ __launch_bounds__(256) void row_normalize(
    const float* __restrict__ out1, float* __restrict__ ext) {
  const int row = blockIdx.x;
  const int tid = threadIdx.x;
  __shared__ float ssum[256];
  __shared__ int   sany[256];

  float* hrow = ext + (size_t)row * (NCOLS + 1);
  const float* orow = out1 + (size_t)row * NCOLS;

  float s = 0.0f;
  int any = 0;
  for (int j = tid; j < NCOLS; j += 256) {
    s += hrow[j];
    any |= (orow[j] > 0.0f) ? 1 : 0;
  }
  ssum[tid] = s;
  sany[tid] = any;
  __syncthreads();
#pragma unroll
  for (int off = 128; off >= 1; off >>= 1) {
    if (tid < off) { ssum[tid] += ssum[tid + off]; sany[tid] |= sany[tid + off]; }
    __syncthreads();
  }
  const float tot = ssum[0];
  const int anyp  = sany[0];
  const float scale = 1.0f / ((tot <= EPSV) ? EPSV : tot);

  if (anyp) {
    for (int j = tid; j < NCOLS; j += 256) hrow[j] *= scale;
  } else {
    for (int j = tid; j < NCOLS; j += 256) hrow[j] = 0.0f;
  }
  if (tid == 0) hrow[NCOLS] = anyp ? 0.0f : 1.0f;  // supple flag column
}

// ---------------------------------------------------------------------------
// Launcher
// ---------------------------------------------------------------------------
extern "C" void kernel_launch(void* const* d_in, const int* in_sizes, int n_in,
                              void* d_out, int out_size, void* d_ws, size_t ws_size,
                              hipStream_t stream) {
  const float* emb  = (const float*)d_in[0];   // 8192 x 512 f32
  const int*   Wmat = (const int*)d_in[1];     // 8192 x 8192 i32
  (void)in_sizes; (void)n_in; (void)out_size; (void)ws_size;

  float* out1 = (float*)d_out;                              // 4096 x 4096
  float* ext  = (float*)d_out + (size_t)LEN * NCOLS;        // 4096 x 4097

  _Float16* ne = (_Float16*)d_ws;                           // 8192 x 512 f16 (8 MB)

  normalize_to_f16<<<N_TOTAL / 8, 256, 0, stream>>>(emb, ne);
  gemm_wmma_epilogue<<<dim3(NCOLS / 64, LEN / 128), 256, 0, stream>>>(ne, Wmat, out1, ext);
  row_normalize<<<LEN, 256, 0, stream>>>(out1, ext);
}
